// SelfAttention_17557826306181
// MI455X (gfx1250) — matrix operand order repaired and checked by reading, not compiled, not run on hardware
//
#include <hip/hip_runtime.h>
#include <stdint.h>

#define B_  4
#define C_  256
#define N_  4096
#define QK_ 32
#define LOG2E 1.4426950408889634f

typedef __attribute__((ext_vector_type(16))) __bf16       v16bf;
typedef __attribute__((ext_vector_type(8)))  float        v8f;
typedef __attribute__((ext_vector_type(8)))  unsigned int v8u;
typedef __attribute__((ext_vector_type(4)))  unsigned int v4u;

static __device__ __forceinline__ unsigned short bf16_bits(float f) {
  __bf16 h = (__bf16)f;
  return __builtin_bit_cast(unsigned short, h);
}

static __device__ __forceinline__ v8f wmma_bf16(v16bf a, v16bf b, v8f c) {
  return __builtin_amdgcn_wmma_f32_16x16x32_bf16(false, a, false, b, (short)0, c,
                                                 false, false);
}

// ---------------------------------------------------------------------------
// Kernel 1: channel-mixing projection  out[b,o,n] = sum_c W[o,c] * x[b,c,n] + bias[o]
// One block per (b, 16-wide n tile); one wave per 16-row o tile.
// x tile staged into LDS transposed (bf16) so the B operand is one contiguous
// 32B ds load per lane per K-step.
// ---------------------------------------------------------------------------
__global__ void proj_kernel(const float* __restrict__ x,
                            const float* __restrict__ W,
                            const float* __restrict__ bias,
                            unsigned short* __restrict__ out, int O) {
  __shared__ __align__(32) unsigned short xT[16][C_];  // [n_local][c] bf16 bits

  const int tid  = threadIdx.x;
  const int nthr = blockDim.x;
  const int blk  = blockIdx.x;
  const int b    = blk >> 8;      // 256 n-tiles per batch
  const int nt   = blk & 255;
  const int n0   = nt * 16;

  for (int i = tid; i < 16 * C_; i += nthr) {
    const int c = i >> 4, j = i & 15;
    xT[j][c] = bf16_bits(x[(b * C_ + c) * N_ + n0 + j]);
  }
  __syncthreads();

  const int w  = tid >> 5;
  const int la = tid & 31;
  const int lm = la & 15, lh = la >> 4;
  const int o0 = w * 16;

  v8f acc = {};
  for (int kb = 0; kb < C_ / 32; ++kb) {
    // A operand: W rows, f32 -> bf16.  Lane lm = row; K = kc+0..15
    const int kc = kb * 32 + lh * 16;
    const float* wr = W + (o0 + lm) * C_ + kc;
    v16bf a;
#pragma unroll
    for (int t = 0; t < 8; ++t) a[t] = (__bf16)wr[t];
#pragma unroll
    for (int t = 0; t < 8; ++t) a[8 + t] = (__bf16)wr[8 + t];

    // B operand from LDS: lane lm = column n, 16 consecutive K halves
    v8u bv = *(const v8u*)&xT[lm][kb * 32 + lh * 16];
    v16bf bm = __builtin_bit_cast(v16bf, bv);
    acc = wmma_bf16(a, bm, acc);
  }

#pragma unroll
  for (int v = 0; v < 8; ++v) {
    const int row = o0 + v + 8 * lh;         // C/D layout: M = v + 8*(lane/16)
    const float r = acc[v] + bias[row];
    out[(b * O + row) * N_ + n0 + lm] = bf16_bits(r);
  }
}

// ---------------------------------------------------------------------------
// Pack f into exact WMMA A-operand images of f^T tiles (16n x 32k):
// lane<16: row n0+lane, K 0..15 ; lane>=16: K 16..31.
// ---------------------------------------------------------------------------
__global__ void pack_fA_kernel(const unsigned short* __restrict__ f,
                               v8u* __restrict__ fA) {
  const int blk = blockIdx.x, lane = threadIdx.x;
  const int b = blk >> 8, nt = blk & 255;
  const int n = nt * 16 + (lane & 15);
  const int kb0 = (lane >> 4) * 16;
  v8u dw;
#pragma unroll
  for (int j = 0; j < 8; ++j) {
    const int k0 = 2 * j + kb0;
    const unsigned int lo = f[(b * QK_ + k0) * N_ + n];
    const unsigned int hi = f[(b * QK_ + k0 + 1) * N_ + n];
    dw[j] = lo | (hi << 16);
  }
  fA[(b * 256 + nt) * 32 + lane] = dw;
}

// ---------------------------------------------------------------------------
// Pack g into exact WMMA B-operand images (32k x 16m):
// lane<16: col m0+lane, rows 0..15 pair-packed ; lane>=16: rows 16..31.
// ---------------------------------------------------------------------------
__global__ void pack_gB_kernel(const unsigned short* __restrict__ g,
                               v8u* __restrict__ gB) {
  const int blk = blockIdx.x, lane = threadIdx.x;
  const int b = blk >> 8, mt = blk & 255;
  const int m = mt * 16 + (lane & 15);
  const int kb0 = (lane >> 4) * 16;
  v8u dw;
#pragma unroll
  for (int j = 0; j < 8; ++j) {
    const unsigned int lo = g[(b * QK_ + kb0 + 2 * j) * N_ + m];
    const unsigned int hi = g[(b * QK_ + kb0 + 2 * j + 1) * N_ + m];
    dw[j] = lo | (hi << 16);
  }
  gB[(b * 256 + mt) * 32 + lane] = dw;
}

// ---------------------------------------------------------------------------
// Pass 1 of flash softmax (log2 domain): per attention row n compute
//   rowbias[n] = -(rm + log2(rs))  so that  p = exp2(s*log2e + rowbias).
// One wave per 16-row n tile; online rescaled running (max, sum) in exp2
// domain; butterfly reduce across the 16 lanes per row.
// ---------------------------------------------------------------------------
__global__ void softmax_stats_kernel(const v8u* __restrict__ fA,
                                     const v8u* __restrict__ gB,
                                     float* __restrict__ rowbias) {
  const int w = threadIdx.x >> 5, lane = threadIdx.x & 31;
  const int gw = blockIdx.x * (blockDim.x >> 5) + w;
  const int b = gw >> 8, nt = gw & 255;

  const v16bf amat =
      __builtin_bit_cast(v16bf, fA[(b * 256 + nt) * 32 + lane]);

  float rm[8], rs[8];
#pragma unroll
  for (int v = 0; v < 8; ++v) { rm[v] = -1e30f; rs[v] = 0.0f; }

  for (int mt = 0; mt < 256; ++mt) {
    const v16bf bmat =
        __builtin_bit_cast(v16bf, gB[(b * 256 + mt) * 32 + lane]);
    v8f acc = {};
    acc = wmma_bf16(amat, bmat, acc);
#pragma unroll
    for (int v = 0; v < 8; ++v) {
      const float t  = acc[v] * LOG2E;
      const float nm = fmaxf(rm[v], t);
      rs[v] = rs[v] * exp2f(rm[v] - nm) + exp2f(t - nm);
      rm[v] = nm;
    }
  }

#pragma unroll
  for (int v = 0; v < 8; ++v) {
#pragma unroll
    for (int off = 1; off < 16; off <<= 1) {
      const float om = __shfl_xor(rm[v], off, 16);
      const float os = __shfl_xor(rs[v], off, 16);
      const float nm = fmaxf(rm[v], om);
      rs[v] = rs[v] * exp2f(rm[v] - nm) + os * exp2f(om - nm);
      rm[v] = nm;
    }
  }

  if ((lane & 15) == 0) {
#pragma unroll
    for (int v = 0; v < 8; ++v) {
      const int r = nt * 16 + v + 8 * (lane >> 4);
      rowbias[b * N_ + r] = -(rm[v] + __log2f(rs[v]));
    }
  }
}

// ---------------------------------------------------------------------------
// Pass 2: fused attention-apply + output GEMM + residual.
// One block per (b, 32-wide m block); 8 waves.  Waves 0 and 1 each recompute
// one 32n x 16m score sub-tile, apply p = exp2(fma(s, log2e, bias)) with bias
// staged in LDS via ASYNC global->LDS copies, and write bf16 p transposed
// into ping-pong LDS.  All 8 waves then accumulate 2 c-tiles x 2 m-tiles:
//   acc[c,m] += h[c, n0..n0+31] (A, two direct 16B global loads, shared by
//   both m-tiles) * p (B, one 32B ds load per m-tile).
// One barrier per iteration (ping-pong makes the single barrier sufficient).
// ---------------------------------------------------------------------------
__global__ void attn_out_kernel(const unsigned short* __restrict__ h,
                                const v8u* __restrict__ fA,
                                const v8u* __restrict__ gB,
                                const float* __restrict__ rowbias,
                                const float* __restrict__ x,
                                const float* __restrict__ gamma,
                                float* __restrict__ out) {
  __shared__ __align__(32) unsigned int pT[2][2][16][16];  // [buf][mtile][m][n/2]
  __shared__ __align__(16) float biasS[N_];                // rowbias for batch b

  const int blk = blockIdx.x;
  const int b = blk >> 7;            // 128 m-blocks per batch
  const int mb = blk & 127;
  const int mt0 = mb * 2;            // first of two 16-wide m tiles
  const int m0 = mt0 * 16;
  const int w = threadIdx.x >> 5, lane = threadIdx.x & 31;
  const int lm = lane & 15, lh = lane >> 4;

  // Async DMA of the softmax bias table (16 KB, no conversion needed) into
  // LDS: per-lane 16B global->LDS copies tracked by ASYNCcnt, overlapped with
  // the gB operand load below, fenced with s_wait_asynccnt before the barrier.
  for (int i = threadIdx.x * 4; i < N_; i += 256 * 4) {
    const unsigned lds_off = (unsigned)(uintptr_t)&biasS[i];
    const float* gp = rowbias + b * N_ + i;
    asm volatile("global_load_async_to_lds_b128 %0, %1, off"
                 :: "v"(lds_off), "v"(gp)
                 : "memory");
  }

  v16bf gmat = {};
  if (w < 2)
    gmat = __builtin_bit_cast(v16bf, gB[(b * 256 + mt0 + w) * 32 + lane]);

  v8f acc[2][2] = {};   // [c-subtile][m-subtile]

  asm volatile("s_wait_asynccnt 0" ::: "memory");
  __syncthreads();      // biasS ready before first producer iteration

  for (int it = 0; it < N_ / 32; ++it) {
    const int n0 = it * 32;

    if (w < 2) {  // wave w produces p sub-tile for m-tile (mt0 + w)
#pragma unroll
      for (int t = 0; t < 2; ++t) {
        const int nt2 = it * 2 + t;
        const v16bf amat =
            __builtin_bit_cast(v16bf, fA[(b * 256 + nt2) * 32 + lane]);
        v8f s = {};
        s = wmma_bf16(amat, gmat, s);
#pragma unroll
        for (int v = 0; v < 8; v += 2) {
          const int r0 = nt2 * 16 + v + 8 * lh;  // global attn row n
          const float p0 = exp2f(fmaf(s[v],     LOG2E, biasS[r0]));
          const float p1 = exp2f(fmaf(s[v + 1], LOG2E, biasS[r0 + 1]));
          const unsigned int pk =
              (unsigned int)bf16_bits(p0) | ((unsigned int)bf16_bits(p1) << 16);
          const int nl = t * 16 + 8 * lh + v;  // local n within 32 (even)
          pT[it & 1][w][lm][nl >> 1] = pk;
        }
      }
    }
    __syncthreads();

    // B operands: lane lm = column m, 16 consecutive n halves from LDS
    const v16bf bm0 =
        __builtin_bit_cast(v16bf, *(const v8u*)&pT[it & 1][0][lm][lh * 8]);
    const v16bf bm1 =
        __builtin_bit_cast(v16bf, *(const v8u*)&pT[it & 1][1][lm][lh * 8]);

#pragma unroll
    for (int tt = 0; tt < 2; ++tt) {
      const int c = (w * 2 + tt) * 16 + lm;
      const unsigned short* hr = h + (b * C_ + c) * N_ + n0 + lh * 16;
      const v4u a_lo = *(const v4u*)hr;          // K = n0+lh*16 .. +7
      const v4u a_hi = *(const v4u*)(hr + 8);    // K = n0+lh*16+8 .. +15
      __builtin_prefetch(hr + 64, 0, 3);         // pull next h lines toward WGP
      v8u av;
#pragma unroll
      for (int q = 0; q < 4; ++q) { av[q] = a_lo[q]; av[4 + q] = a_hi[q]; }
      const v16bf am = __builtin_bit_cast(v16bf, av);
      acc[tt][0] = wmma_bf16(am, bm0, acc[tt][0]);
      acc[tt][1] = wmma_bf16(am, bm1, acc[tt][1]);
    }
  }

  const float ga = gamma[0];
#pragma unroll
  for (int tt = 0; tt < 2; ++tt) {
#pragma unroll
    for (int mm = 0; mm < 2; ++mm) {
#pragma unroll
      for (int v = 0; v < 8; ++v) {
        const int c = (w * 2 + tt) * 16 + v + 8 * lh;
        const int idx = (b * C_ + c) * N_ + m0 + mm * 16 + lm;
        out[idx] = ga * acc[tt][mm][v] + x[idx];
      }
    }
  }
}

// ---------------------------------------------------------------------------
extern "C" void kernel_launch(void* const* d_in, const int* in_sizes, int n_in,
                              void* d_out, int out_size, void* d_ws,
                              size_t ws_size, hipStream_t stream) {
  const float* x     = (const float*)d_in[0];
  const float* Wf    = (const float*)d_in[1];
  const float* bf    = (const float*)d_in[2];
  const float* Wg    = (const float*)d_in[3];
  const float* bg    = (const float*)d_in[4];
  const float* Wh    = (const float*)d_in[5];
  const float* bh    = (const float*)d_in[6];
  const float* gamma = (const float*)d_in[7];
  float* out = (float*)d_out;

  char* ws = (char*)d_ws;
  size_t off = 0;
  auto carve = [&](size_t bytes) -> void* {
    void* p = ws + off;
    off = (off + bytes + 255) & ~(size_t)255;
    return p;
  };
  unsigned short* fbuf = (unsigned short*)carve((size_t)B_ * QK_ * N_ * 2);
  unsigned short* gbuf = (unsigned short*)carve((size_t)B_ * QK_ * N_ * 2);
  unsigned short* hbuf = (unsigned short*)carve((size_t)B_ * C_ * N_ * 2);
  v8u* fA = (v8u*)carve((size_t)B_ * 256 * 32 * sizeof(v8u));
  v8u* gB = (v8u*)carve((size_t)B_ * 256 * 32 * sizeof(v8u));
  float* rowbias = (float*)carve((size_t)B_ * N_ * 4);
  (void)ws_size; (void)in_sizes; (void)n_in; (void)out_size;

  const int nblk16 = B_ * (N_ / 16);  // 1024

  proj_kernel<<<nblk16, 64, 0, stream>>>(x, Wf, bf, fbuf, QK_);
  proj_kernel<<<nblk16, 64, 0, stream>>>(x, Wg, bg, gbuf, QK_);
  proj_kernel<<<nblk16, 512, 0, stream>>>(x, Wh, bh, hbuf, C_);

  pack_fA_kernel<<<nblk16, 32, 0, stream>>>(fbuf, fA);
  pack_gB_kernel<<<nblk16, 32, 0, stream>>>(gbuf, gB);

  softmax_stats_kernel<<<nblk16 / 8, 256, 0, stream>>>(fA, gB, rowbias);

  attn_out_kernel<<<B_ * (N_ / 32), 256, 0, stream>>>(hbuf, fA, gB, rowbias, x,
                                                      gamma, out);
}
